// DistributedMoERouter_65446711656460
// MI455X (gfx1250) — matrix operand order repaired and checked by reading, not compiled, not run on hardware
//
#include <hip/hip_runtime.h>
#include <hip/hip_bf16.h>

typedef __attribute__((ext_vector_type(16))) _Float16 v16h;
typedef __attribute__((ext_vector_type(8)))  float    v8f;
typedef __attribute__((ext_vector_type(4)))  float    v4f;

#define D_MODEL     4096
#define NUM_EXPERTS 32

// ---------------------------------------------------------------------------
// Kernel 0: zero d_out and per-expert counters (harness poisons with 0xAA).
// ---------------------------------------------------------------------------
__global__ void moe_zero_kernel(float* __restrict__ out, long long n,
                                int* __restrict__ counts) {
  long long i = (long long)blockIdx.x * blockDim.x + threadIdx.x;
  long long stride = (long long)gridDim.x * blockDim.x;
  for (long long j = i; j < n; j += stride) out[j] = 0.0f;
  if (i < NUM_EXPERTS) counts[i] = 0;
}

// ---------------------------------------------------------------------------
// Kernel 1: gating. One wave32 per token; lane e computes logit for expert e.
// Wave-wide top-2 (with smaller-index tie-break, matching jax top_k), softmax
// over the two values, then lane 0 scatter-appends into per-expert buckets.
// ---------------------------------------------------------------------------
__global__ __launch_bounds__(256)
void moe_gate_kernel(const float* __restrict__ x,
                     const float* __restrict__ gw,
                     int* __restrict__ counts,
                     int* __restrict__ btok,
                     float* __restrict__ bwt,
                     int T) {
  const int wave = threadIdx.x >> 5;
  const int lane = threadIdx.x & 31;
  const int t = blockIdx.x * (blockDim.x >> 5) + wave;
  if (t >= T) return;

  const float* xr = x + (size_t)t * D_MODEL;      // broadcast across lanes
  const float* gr = gw + (size_t)lane * D_MODEL;  // one expert row per lane

  float acc = 0.0f;
  for (int d = 0; d < D_MODEL; d += 4) {
    v4f xv = *(const v4f*)(xr + d);
    v4f gv = *(const v4f*)(gr + d);
    acc += xv.x * gv.x + xv.y * gv.y + xv.z * gv.z + xv.w * gv.w;
  }

  // top-1 (value, index) butterfly reduction; prefer smaller index on ties
  float v1 = acc; int i1 = lane;
  for (int off = 16; off >= 1; off >>= 1) {
    float ov = __shfl_xor(v1, off, 32);
    int   oi = __shfl_xor(i1, off, 32);
    if (ov > v1 || (ov == v1 && oi < i1)) { v1 = ov; i1 = oi; }
  }
  // top-2: mask out the winner, reduce again
  float m2 = (lane == i1) ? -3.0e38f : acc;
  float v2 = m2; int i2 = lane;
  for (int off = 16; off >= 1; off >>= 1) {
    float ov = __shfl_xor(v2, off, 32);
    int   oi = __shfl_xor(i2, off, 32);
    if (ov > v2 || (ov == v2 && oi < i2)) { v2 = ov; i2 = oi; }
  }

  // softmax over {v1, v2} (v1 >= v2)
  const float e2  = __expf(v2 - v1);
  const float inv = 1.0f / (1.0f + e2);
  const float w1  = inv;
  const float w2  = e2 * inv;

  if (lane == 0) {
    int p1 = atomicAdd(&counts[i1], 1);
    btok[i1 * T + p1] = t;
    bwt [i1 * T + p1] = w1;
    int p2 = atomicAdd(&counts[i2], 1);
    btok[i2 * T + p2] = t;
    bwt [i2 * T + p2] = w2;
  }
}

// ---------------------------------------------------------------------------
// Kernel 2: grouped GEMM per expert with f16 WMMA (f32 accumulate).
// Block = 8 waves; block tile = 16 tokens x 128 columns; each wave does a
// 16x16 WMMA tile over K=4096 in steps of 32. fp32 operands are loaded with
// b128 loads and converted to f16 in-register (weights stream once: 2.15 GB).
// ---------------------------------------------------------------------------
__device__ inline void pack8(v16h& dst, v4f lo, v4f hi, int base) {
  dst[base + 0] = (_Float16)lo.x; dst[base + 1] = (_Float16)lo.y;
  dst[base + 2] = (_Float16)lo.z; dst[base + 3] = (_Float16)lo.w;
  dst[base + 4] = (_Float16)hi.x; dst[base + 5] = (_Float16)hi.y;
  dst[base + 6] = (_Float16)hi.z; dst[base + 7] = (_Float16)hi.w;
}

__global__ __launch_bounds__(256)
void moe_expert_gemm_kernel(const float* __restrict__ x,
                            const float* __restrict__ ew,
                            const float* __restrict__ eb,
                            const int*   __restrict__ counts,
                            const int*   __restrict__ btok,
                            const float* __restrict__ bwt,
                            float* __restrict__ out,
                            int T) {
  const int e   = blockIdx.y;
  const int cnt = counts[e];
  if (cnt == 0) return;

  const int wave = threadIdx.x >> 5;
  const int lane = threadIdx.x & 31;
  const int half = lane >> 4;   // 0: lanes 0-15, 1: lanes 16-31
  const int sub  = lane & 15;

  // Output column owned by this lane (B matrix N index, C/D N index).
  const int n = blockIdx.x * 128 + wave * 16 + sub;
  const float* Wrow = ew + ((size_t)e * D_MODEL + n) * D_MODEL;  // row n of W_e
  const float  bias = eb[(size_t)e * D_MODEL + n];

  const int*   etok = btok + (size_t)e * T;
  const float* ewt  = bwt  + (size_t)e * T;

  const int mtiles = (cnt + 15) >> 4;
  for (int mt = 0; mt < mtiles; ++mt) {
    // A-operand row for this lane (M = lane&15). Out-of-range rows read
    // token 0 (safe); their results are never stored.
    const int arow = mt * 16 + sub;
    const int tA   = etok[arow < cnt ? arow : 0];
    const float* xrow = x + (size_t)tA * D_MODEL;

    v8f c = {0.f, 0.f, 0.f, 0.f, 0.f, 0.f, 0.f, 0.f};

#pragma unroll 2
    for (int k0 = 0; k0 < D_MODEL; k0 += 32) {
      // A (16x32): lanes 0-15 hold K=0..15, lanes 16-31 hold K=16..31
      //            (same K order as B).
      const int ka = k0 + half * 16;
      v4f a0 = *(const v4f*)(xrow + ka);
      v4f a1 = *(const v4f*)(xrow + ka + 4);
      v4f a2 = *(const v4f*)(xrow + ka + 8);
      v4f a3 = *(const v4f*)(xrow + ka + 12);
      // B (32x16): lanes 0-15 hold K=0..15, lanes 16-31 hold K=16..31.
      const int kb = k0 + half * 16;
      v4f b0 = *(const v4f*)(Wrow + kb);
      v4f b1 = *(const v4f*)(Wrow + kb + 4);
      v4f b2 = *(const v4f*)(Wrow + kb + 8);
      v4f b3 = *(const v4f*)(Wrow + kb + 12);

      v16h a, b;
      pack8(a, a0, a1, 0); pack8(a, a2, a3, 8);
      pack8(b, b0, b1, 0); pack8(b, b2, b3, 8);

      // (neg_a, A, neg_b, B, c_mod, C, reuse_a, reuse_b)
      c = __builtin_amdgcn_wmma_f32_16x16x32_f16(
              false, a, false, b, (short)0, c, false, false);
    }

    // Epilogue: C/D layout — lane l, vgpr i => M = i + 8*(l>>4), N = l&15.
    // out[token] += w * (y + bias); each out element receives exactly two
    // fp adds (one per routed expert), so atomic order does not change bits.
#pragma unroll
    for (int i = 0; i < 8; ++i) {
      const int row = mt * 16 + i + half * 8;
      if (row < cnt) {
        const int   t = etok[row];
        const float w = ewt[row];
        float* dst = out + (size_t)t * D_MODEL + n;
        __hip_atomic_fetch_add(dst, w * (c[i] + bias),
                               __ATOMIC_RELAXED, __HIP_MEMORY_SCOPE_AGENT);
      }
    }
  }
}

// ---------------------------------------------------------------------------
// Launch. Inputs: x [2,2048,4096] f32, gate_w [32,4096] f32,
// expert_w [32,4096,4096] f32, expert_b [32,4096] f32. Output f32 [2,2048,4096].
// Workspace: counts (64 ints, padded) + btok (E*T ints) + bwt (E*T floats).
// ---------------------------------------------------------------------------
extern "C" void kernel_launch(void* const* d_in, const int* in_sizes, int n_in,
                              void* d_out, int out_size, void* d_ws, size_t ws_size,
                              hipStream_t stream) {
  const float* x  = (const float*)d_in[0];
  const float* gw = (const float*)d_in[1];
  const float* ew = (const float*)d_in[2];
  const float* eb = (const float*)d_in[3];
  float* out = (float*)d_out;

  const int T = in_sizes[0] / D_MODEL;  // 4096 tokens

  int*   counts = (int*)d_ws;
  int*   btok   = counts + 64;                     // 256 B offset
  float* bwt    = (float*)(btok + NUM_EXPERTS * T);

  const long long outN = (long long)T * D_MODEL;
  moe_zero_kernel<<<4096, 256, 0, stream>>>(out, outN, counts);

  moe_gate_kernel<<<(T + 7) / 8, 256, 0, stream>>>(x, gw, counts, btok, bwt, T);

  dim3 grid(D_MODEL / 128, NUM_EXPERTS);
  moe_expert_gemm_kernel<<<grid, 256, 0, stream>>>(x, ew, eb, counts, btok, bwt,
                                                   out, T);
}
